// GraphormerBlock_64330020159495
// MI455X (gfx1250) — matrix operand order repaired and checked by reading, not compiled, not run on hardware
//
#include <hip/hip_runtime.h>
#include <hip/hip_bf16.h>
#include <math.h>

// ---------------- problem constants (match reference) ----------------
#define Nn 2048
#define Cc 256
#define Hh 8
#define Dd 32
#define Tt 8
#define Rr 12
#define Ee 32768
#define Ff 1024          // 4*C
#define BUCKETS 21
#define ECAP 2048        // LDS edge-list capacity per 16-row query block (avg ~256)

typedef __attribute__((ext_vector_type(16))) _Float16 v16h;
typedef __attribute__((ext_vector_type(8)))  _Float16 v8h;
typedef __attribute__((ext_vector_type(8)))  float    v8f;

// ---------------- WMMA fragment helpers (CDNA5 wave32 layouts) ----------------
// A (16x32 f16): lane m (m<16) holds K = {0..7} U {16..23} of row m,
//                lane m+16 holds K = {8..15} U {24..31} of row m.
__device__ __forceinline__ v16h ld_a(const _Float16* __restrict__ base, int lda) {
  int l = threadIdx.x & 31;
  const _Float16* p = base + (l & 15) * lda + ((l >> 4) * 8);
  v8h lo = *(const v8h*)p;
  v8h hi = *(const v8h*)(p + 16);
  return __builtin_shufflevector(lo, hi, 0,1,2,3,4,5,6,7,8,9,10,11,12,13,14,15);
}
// B (32x16 f16), source stored column-contiguous: element (k,n) at base[n*ldb + k].
// lane n holds K = {0..7} U {16..23} of column n; lane n+16 holds K = {8..15} U {24..31}.
__device__ __forceinline__ v16h ld_b(const _Float16* __restrict__ base, int ldb) {
  int l = threadIdx.x & 31;
  const _Float16* p = base + (l & 15) * ldb + ((l >> 4) * 8);
  v8h lo = *(const v8h*)p;
  v8h hi = *(const v8h*)(p + 16);
  return __builtin_shufflevector(lo, hi, 0,1,2,3,4,5,6,7,8,9,10,11,12,13,14,15);
}
__device__ __forceinline__ v8f wmma_f16(v16h a, v16h b, v8f c) {
  return __builtin_amdgcn_wmma_f32_16x16x32_f16(false, a, false, b, (short)0, c, false, false);
}
// C/D (16x16 f32): VGPR j -> row j (lanes 0-15) / row j+8 (lanes 16-31), col = lane&15.

// ---------------- prep kernels ----------------
__global__ void wt_kernel(const float* __restrict__ w, _Float16* __restrict__ wt, int K, int Ncol) {
  int i = blockIdx.x * blockDim.x + threadIdx.x;
  if (i < K * Ncol) { int k = i / Ncol, n = i % Ncol; wt[n * K + k] = (_Float16)w[i]; }
}
__global__ void zero_i32(int* __restrict__ p, int n) {
  int i = blockIdx.x * blockDim.x + threadIdx.x;
  if (i < n) p[i] = 0;
}
__global__ void count_kernel(const int* __restrict__ src, int* __restrict__ cnt, int e) {
  int i = blockIdx.x * blockDim.x + threadIdx.x;
  if (i < e) atomicAdd(&cnt[src[i]], 1);
}
__global__ void scan_kernel(const int* __restrict__ cnt, int* __restrict__ off, int n) {
  if (blockIdx.x == 0 && threadIdx.x == 0) {
    int s = 0;
    for (int i = 0; i < n; ++i) { off[i] = s; s += cnt[i]; }
    off[n] = s;
  }
}
__global__ void scatter_kernel(const int* __restrict__ src, const int* __restrict__ dst,
                               const int* __restrict__ rel, const int* __restrict__ off,
                               int* __restrict__ cur, unsigned* __restrict__ ent, int e) {
  int i = blockIdx.x * blockDim.x + threadIdx.x;
  if (i < e) {
    int s = src[i];
    int p = off[s] + atomicAdd(&cur[s], 1);
    ent[p] = (unsigned)dst[i] | ((unsigned)rel[i] << 11);   // dst:11 bits, rel:4 bits
  }
}

// ---------------- LayerNorm: f32 row -> f16 row ----------------
__global__ __launch_bounds__(256) void ln_kernel(const float* __restrict__ x,
                                                 const float* __restrict__ g,
                                                 const float* __restrict__ b,
                                                 _Float16* __restrict__ out) {
  __shared__ float rs[256], rq[256];
  int row = blockIdx.x, t = threadIdx.x;
  float v = x[row * Cc + t];
  rs[t] = v; rq[t] = v * v;
  __syncthreads();
  for (int s = 128; s > 0; s >>= 1) {
    if (t < s) { rs[t] += rs[t + s]; rq[t] += rq[t + s]; }
    __syncthreads();
  }
  float mu  = rs[0] * (1.f / Cc);
  float var = rq[0] * (1.f / Cc) - mu * mu;
  float inv = rsqrtf(var + 1e-5f);
  out[row * Cc + t] = (_Float16)((v - mu) * inv * g[t] + b[t]);
}

// ---------------- fused QKV GEMM (A tile reused 3x) ----------------
__global__ __launch_bounds__(128) void qkv_kernel(const _Float16* __restrict__ h,
    const _Float16* __restrict__ wqt, const _Float16* __restrict__ wkt, const _Float16* __restrict__ wvt,
    const float* __restrict__ bq, const float* __restrict__ bk, const float* __restrict__ bv,
    _Float16* __restrict__ q, _Float16* __restrict__ k, _Float16* __restrict__ vt) {
  int n0 = blockIdx.x * 16;
  int w = threadIdx.x >> 5, l = threadIdx.x & 31;
  int c0 = blockIdx.y * 64 + w * 16;
  v8f aq = {}, ak = {}, av = {};
  for (int kk = 0; kk < Cc; kk += 32) {
    v16h a = ld_a(h + n0 * Cc + kk, Cc);
    aq = wmma_f16(a, ld_b(wqt + c0 * Cc + kk, Cc), aq);
    ak = wmma_f16(a, ld_b(wkt + c0 * Cc + kk, Cc), ak);
    av = wmma_f16(a, ld_b(wvt + c0 * Cc + kk, Cc), av);
  }
  int col = c0 + (l & 15);
  float bqv = bq[col], bkv = bk[col], bvv = bv[col];
#pragma unroll
  for (int j = 0; j < 8; ++j) {
    int row = n0 + j + ((l >> 4) * 8);
    q[row * Cc + col] = (_Float16)(aq[j] + bqv);
    k[row * Cc + col] = (_Float16)(ak[j] + bkv);
    vt[col * Nn + row] = (_Float16)(av[j] + bvv);   // V stored transposed for attn@V B-operand
  }
}

// ---------------- fused flash attention with on-the-fly structural bias ----------------
// grid: (N/16, H/4); block: 128 (4 waves). Wave w handles head blockIdx.y*4+w
// for the 16 query rows. Bias is never materialized: typepair/temporal recomputed,
// edges pulled from CSR into an LDS list once per query block.
__global__ __launch_bounds__(128) void attn_kernel(
    const _Float16* __restrict__ qf, const _Float16* __restrict__ kf, const _Float16* __restrict__ vt,
    const int* __restrict__ tok, const float* __restrict__ tvec, const int* __restrict__ seedp,
    const float* __restrict__ adjb,   // [R,H]
    const float* __restrict__ tpb,    // [T,T,H]
    const float* __restrict__ tmb,    // [BUCKETS,H]
    const int* __restrict__ coff, const unsigned* __restrict__ cent,
    _Float16* __restrict__ ofb) {
  __shared__ __align__(32) float    S[4][16][64];
  __shared__ __align__(32) _Float16 P[4][16][64];
  __shared__ float rm[4][16], rl[4][16], rscale[4][16];
  __shared__ unsigned elist[ECAP];
  __shared__ int ecnt;
  __shared__ int   ttr[16];
  __shared__ float tr[16];

  const int n0   = blockIdx.x * 16;
  const int w    = threadIdx.x >> 5;
  const int l    = threadIdx.x & 31;
  const int head = blockIdx.y * 4 + w;
  const int seed = *seedp;

  if (threadIdx.x == 0) ecnt = 0;
  if (threadIdx.x < 16) {
    ttr[threadIdx.x] = tok[n0 + threadIdx.x];
    tr[threadIdx.x]  = tvec[n0 + threadIdx.x];
  }
  if (l < 16) { rm[w][l] = -1e30f; rl[w][l] = 0.f; }
  __syncthreads();

  // gather this block's edges (row-local id packed into bits 15..18)
  for (int r = 0; r < 16; ++r) {
    int beg = coff[n0 + r], end = coff[n0 + r + 1];
    for (int e = beg + (int)threadIdx.x; e < end; e += 128) {
      int slot = atomicAdd(&ecnt, 1);
      if (slot < ECAP) elist[slot] = cent[e] | ((unsigned)r << 15);
    }
  }
  __syncthreads();
  const int ne = min(ecnt, ECAP);

  const v16h aq = ld_a(qf + n0 * Cc + head * Dd, Cc);   // Q tile, K=32 = one fragment
  v8f o0 = {}, o1 = {};
  const float scl = 0.17677669529663687f;               // 1/sqrt(D)

  for (int m0 = 0; m0 < Nn; m0 += 64) {
    // ---- scores for 64 keys: 4 WMMAs, spill to LDS scaled by 1/sqrt(D)
#pragma unroll
    for (int t = 0; t < 4; ++t) {
      v16h bk = ld_b(kf + (m0 + t * 16) * Cc + head * Dd, Cc);  // B = K^T, rows contiguous
      v8f z = {};
      v8f s = wmma_f16(aq, bk, z);
#pragma unroll
      for (int j = 0; j < 8; ++j)
        S[w][j + ((l >> 4) * 8)][t * 16 + (l & 15)] = s[j] * scl;
    }
    __syncthreads();

    // ---- dense bias: typepair + temporal (first `seed` rows)
    for (int i = l; i < 16 * 64; i += 32) {
      int r = i >> 6, c = i & 63, m = m0 + c;
      float bia = tpb[(ttr[r] * Tt + tok[m]) * Hh + head];
      if (n0 + r < seed) {
        float dt = tvec[m] - tr[r];
        float sl = copysignf(log1pf(fabsf(dt) + 1e-6f), dt);
        float cl = fminf(fmaxf(sl, -5.f), 5.f);
        float nr = (cl + 5.f) / (10.f + 1e-9f);
        int idx = (int)floorf(nr * (BUCKETS - 1));
        idx = max(0, min(BUCKETS - 1, idx));
        bia += tmb[idx * Hh + head];
      }
      S[w][r][c] += bia;
    }
    // ---- sparse edge bias (duplicates allowed -> LDS atomic f32 add)
    for (int i = l; i < ne; i += 32) {
      unsigned ent = elist[i];
      int dst = (int)(ent & 2047u);
      if (dst >= m0 && dst < m0 + 64) {
        int rel = (int)((ent >> 11) & 15u);
        int r   = (int)((ent >> 15) & 15u);
        atomicAdd(&S[w][r][dst - m0], adjb[rel * Hh + head]);
      }
    }
    __syncthreads();

    // ---- online softmax over the chunk (2 lanes per row)
    {
      int r = l & 15, hhf = l >> 4;
      float cm = -1e30f;
#pragma unroll
      for (int c = 0; c < 32; ++c) cm = fmaxf(cm, S[w][r][hhf * 32 + c]);
      cm = fmaxf(cm, __shfl_xor(cm, 16));
      float mo = rm[w][r];
      float mn = fmaxf(mo, cm);
      float cs = 0.f;
#pragma unroll
      for (int c = 0; c < 32; ++c) {
        float p = __expf(S[w][r][hhf * 32 + c] - mn);
        cs += p;
        P[w][r][hhf * 32 + c] = (_Float16)p;
      }
      cs += __shfl_xor(cs, 16);
      if (l < 16) {
        float sc2 = __expf(mo - mn);
        rscale[w][r] = sc2;
        rl[w][r] = rl[w][r] * sc2 + cs;
        rm[w][r] = mn;
      }
    }
    __syncthreads();

    // ---- rescale running O, then accumulate P @ V (4 WMMAs)
#pragma unroll
    for (int j = 0; j < 8; ++j) {
      float sc2 = rscale[w][j + ((l >> 4) * 8)];
      o0[j] *= sc2; o1[j] *= sc2;
    }
#pragma unroll
    for (int ks = 0; ks < 2; ++ks) {
      const _Float16* pr = &P[w][l & 15][ks * 32 + ((l >> 4) * 8)];
      v8h lo = *(const v8h*)pr;
      v8h hi = *(const v8h*)(pr + 16);
      v16h ap = __builtin_shufflevector(lo, hi, 0,1,2,3,4,5,6,7,8,9,10,11,12,13,14,15);
      o0 = wmma_f16(ap, ld_b(vt + (head * Dd +  0) * Nn + m0 + ks * 32, Nn), o0);
      o1 = wmma_f16(ap, ld_b(vt + (head * Dd + 16) * Nn + m0 + ks * 32, Nn), o1);
    }
    __syncthreads();
  }

  // ---- normalize + store f16
#pragma unroll
  for (int j = 0; j < 8; ++j) {
    int r = j + ((l >> 4) * 8);
    float inv = 1.f / rl[w][r];
    int row = n0 + r, col = head * Dd + (l & 15);
    ofb[row * Cc + col]      = (_Float16)(o0[j] * inv);
    ofb[row * Cc + col + 16] = (_Float16)(o1[j] * inv);
  }
}

// ---------------- o@Wo + bo + X -> x (f32) ----------------
__global__ __launch_bounds__(128) void xo_kernel(const _Float16* __restrict__ o,
    const _Float16* __restrict__ wot, const float* __restrict__ bo,
    const float* __restrict__ X, float* __restrict__ x) {
  int n0 = blockIdx.x * 16;
  int w = threadIdx.x >> 5, l = threadIdx.x & 31;
  int c0 = blockIdx.y * 64 + w * 16;
  v8f acc = {};
  for (int kk = 0; kk < Cc; kk += 32)
    acc = wmma_f16(ld_a(o + n0 * Cc + kk, Cc), ld_b(wot + c0 * Cc + kk, Cc), acc);
  int col = c0 + (l & 15);
  float bb = bo[col];
#pragma unroll
  for (int j = 0; j < 8; ++j) {
    int row = n0 + j + ((l >> 4) * 8);
    x[row * Cc + col] = X[row * Cc + col] + acc[j] + bb;
  }
}

// ---------------- FFN1: h2@W1 + b1 -> exact GELU -> f16 ----------------
__global__ __launch_bounds__(128) void ffn1_kernel(const _Float16* __restrict__ h2,
    const _Float16* __restrict__ w1t, const float* __restrict__ b1, _Float16* __restrict__ g) {
  int n0 = blockIdx.x * 16;
  int w = threadIdx.x >> 5, l = threadIdx.x & 31;
  int c0 = blockIdx.y * 64 + w * 16;
  v8f acc = {};
  for (int kk = 0; kk < Cc; kk += 32)
    acc = wmma_f16(ld_a(h2 + n0 * Cc + kk, Cc), ld_b(w1t + c0 * Cc + kk, Cc), acc);
  int col = c0 + (l & 15);
  float bb = b1[col];
#pragma unroll
  for (int j = 0; j < 8; ++j) {
    int row = n0 + j + ((l >> 4) * 8);
    float t = acc[j] + bb;
    float ge = 0.5f * t * (1.f + erff(t * 0.70710678118654752f));
    g[row * Ff + col] = (_Float16)ge;
  }
}

// ---------------- FFN2: g@W2 + b2 + x -> d_out (f32) ----------------
__global__ __launch_bounds__(128) void ffn2_kernel(const _Float16* __restrict__ g,
    const _Float16* __restrict__ w2t, const float* __restrict__ b2,
    const float* __restrict__ x, float* __restrict__ out) {
  int n0 = blockIdx.x * 16;
  int w = threadIdx.x >> 5, l = threadIdx.x & 31;
  int c0 = blockIdx.y * 64 + w * 16;
  v8f acc = {};
  for (int kk = 0; kk < Ff; kk += 32)
    acc = wmma_f16(ld_a(g + n0 * Ff + kk, Ff), ld_b(w2t + c0 * Ff + kk, Ff), acc);
  int col = c0 + (l & 15);
  float bb = b2[col];
#pragma unroll
  for (int j = 0; j < 8; ++j) {
    int row = n0 + j + ((l >> 4) * 8);
    out[row * Cc + col] = x[row * Cc + col] + acc[j] + bb;
  }
}

// ---------------- launch ----------------
extern "C" void kernel_launch(void* const* d_in, const int* in_sizes, int n_in,
                              void* d_out, int out_size, void* d_ws, size_t ws_size,
                              hipStream_t stream) {
  (void)in_sizes; (void)n_in; (void)out_size; (void)ws_size;

  const float* X    = (const float*)d_in[0];
  const int*   tok  = (const int*)  d_in[1];
  const int*   esrc = (const int*)  d_in[2];
  const int*   edst = (const int*)  d_in[3];
  const int*   erel = (const int*)  d_in[4];
  const float* tvec = (const float*)d_in[5];
  const int*   seed = (const int*)  d_in[6];
  const float* Wq = (const float*)d_in[7];  const float* bq = (const float*)d_in[8];
  const float* Wk = (const float*)d_in[9];  const float* bk = (const float*)d_in[10];
  const float* Wv = (const float*)d_in[11]; const float* bv = (const float*)d_in[12];
  const float* Wo = (const float*)d_in[13]; const float* bo = (const float*)d_in[14];
  const float* l1g = (const float*)d_in[15]; const float* l1b = (const float*)d_in[16];
  const float* l2g = (const float*)d_in[17]; const float* l2b = (const float*)d_in[18];
  const float* W1 = (const float*)d_in[19]; const float* b1 = (const float*)d_in[20];
  const float* W2 = (const float*)d_in[21]; const float* b2 = (const float*)d_in[22];
  const float* adjb = (const float*)d_in[23];
  const float* tpb  = (const float*)d_in[24];
  const float* tmb  = (const float*)d_in[25];
  float* out = (float*)d_out;

  // workspace layout (all 256B aligned)
  char* ws = (char*)d_ws;
  size_t o_wqt = 0;
  size_t o_wkt = o_wqt + 131072;
  size_t o_wvt = o_wkt + 131072;
  size_t o_wot = o_wvt + 131072;
  size_t o_w1t = o_wot + 131072;
  size_t o_w2t = o_w1t + 524288;
  size_t o_h1  = o_w2t + 524288;
  size_t o_q   = o_h1  + 1048576;
  size_t o_k   = o_q   + 1048576;
  size_t o_vt  = o_k   + 1048576;
  size_t o_of  = o_vt  + 1048576;
  size_t o_x   = o_of  + 1048576;
  size_t o_h2  = o_x   + 2097152;
  size_t o_g   = o_h2  + 1048576;
  size_t o_cnt = o_g   + 4194304;
  size_t o_cur = o_cnt + 8192;
  size_t o_off = o_cur + 8192;
  size_t o_ent = o_off + 8448;

  _Float16* wqt = (_Float16*)(ws + o_wqt);
  _Float16* wkt = (_Float16*)(ws + o_wkt);
  _Float16* wvt = (_Float16*)(ws + o_wvt);
  _Float16* wot = (_Float16*)(ws + o_wot);
  _Float16* w1t = (_Float16*)(ws + o_w1t);
  _Float16* w2t = (_Float16*)(ws + o_w2t);
  _Float16* h1  = (_Float16*)(ws + o_h1);
  _Float16* qf  = (_Float16*)(ws + o_q);
  _Float16* kf  = (_Float16*)(ws + o_k);
  _Float16* vt  = (_Float16*)(ws + o_vt);
  _Float16* of  = (_Float16*)(ws + o_of);
  float*    xb  = (float*)   (ws + o_x);
  _Float16* h2  = (_Float16*)(ws + o_h2);
  _Float16* ga  = (_Float16*)(ws + o_g);
  int*      cnt = (int*)     (ws + o_cnt);
  int*      cur = (int*)     (ws + o_cur);
  int*      cof = (int*)     (ws + o_off);
  unsigned* ent = (unsigned*)(ws + o_ent);

  // 1) weight transpose+convert to f16
  wt_kernel<<<(Cc*Cc + 255)/256, 256, 0, stream>>>(Wq, wqt, Cc, Cc);
  wt_kernel<<<(Cc*Cc + 255)/256, 256, 0, stream>>>(Wk, wkt, Cc, Cc);
  wt_kernel<<<(Cc*Cc + 255)/256, 256, 0, stream>>>(Wv, wvt, Cc, Cc);
  wt_kernel<<<(Cc*Cc + 255)/256, 256, 0, stream>>>(Wo, wot, Cc, Cc);
  wt_kernel<<<(Cc*Ff + 255)/256, 256, 0, stream>>>(W1, w1t, Cc, Ff);
  wt_kernel<<<(Ff*Cc + 255)/256, 256, 0, stream>>>(W2, w2t, Ff, Cc);

  // 2) edge CSR build
  zero_i32<<<(2*Nn + 255)/256, 256, 0, stream>>>(cnt, 2*Nn);   // cnt + cur (contiguous)
  count_kernel<<<(Ee + 255)/256, 256, 0, stream>>>(esrc, cnt, Ee);
  scan_kernel<<<1, 1, 0, stream>>>(cnt, cof, Nn);
  scatter_kernel<<<(Ee + 255)/256, 256, 0, stream>>>(esrc, edst, erel, cof, cur, ent, Ee);

  // 3) LN1 + fused QKV
  ln_kernel<<<Nn, 256, 0, stream>>>(X, l1g, l1b, h1);
  qkv_kernel<<<dim3(Nn/16, Cc/64), 128, 0, stream>>>(h1, wqt, wkt, wvt, bq, bk, bv, qf, kf, vt);

  // 4) fused flash attention with structural bias
  attn_kernel<<<dim3(Nn/16, Hh/4), 128, 0, stream>>>(qf, kf, vt, tok, tvec, seed,
                                                     adjb, tpb, tmb, cof, ent, of);

  // 5) output projection + residual, LN2, FFN
  xo_kernel<<<dim3(Nn/16, Cc/64), 128, 0, stream>>>(of, wot, bo, X, xb);
  ln_kernel<<<Nn, 256, 0, stream>>>(xb, l2g, l2b, h2);
  ffn1_kernel<<<dim3(Nn/16, Ff/64), 128, 0, stream>>>(h2, w1t, b1, ga);
  ffn2_kernel<<<dim3(Nn/16, Cc/64), 128, 0, stream>>>(ga, w2t, b2, xb, out);
}
